// Mamba_49572512530984
// MI455X (gfx1250) — compile-verified, matrix operand order checked by reading, not run on hardware
//
#include <hip/hip_runtime.h>
#include <hip/hip_bf16.h>
#include <stdint.h>

typedef __attribute__((ext_vector_type(16))) _Float16 v16h;
typedef __attribute__((ext_vector_type(8)))  _Float16 v8h;
typedef __attribute__((ext_vector_type(8)))  float    v8f;
typedef __attribute__((ext_vector_type(4)))  float    v4f;

#define BATCH 2
#define SEQ   1024
#define H     128
#define E     256
#define N_ST  128           // state size n == H
#define R     8
#define KCONV 4
#define DBC_N (R + 2 * H)   // 264 (logical)
#define DBC_P 320           // padded to multiple of 64 for WMMA tiling

// ---------------------------------------------------------------------------
// CDNA5 async global->LDS copy (ASYNCcnt-tracked), ISA 08_async_tensor §4.
// vdst VGPR carries the LDS byte address; 64-bit global address in VGPR pair.
// ---------------------------------------------------------------------------
__device__ __forceinline__ void async_load_b128_to_lds(const float* gptr,
                                                       void* lds_ptr) {
  uint32_t lds_off = (uint32_t)(uintptr_t)lds_ptr;   // low 32 bits = LDS offset
  asm volatile("global_load_async_to_lds_b128 %0, %1, off"
               :
               : "v"(lds_off), "v"(gptr)
               : "memory");
}
__device__ __forceinline__ void wait_asynccnt0() {
  asm volatile("s_wait_asynccnt 0" ::: "memory");
}

// ---------------------------------------------------------------------------
// RMSNorm: one wave32 per row of 128 elements (4 per lane), shfl reduction.
// ---------------------------------------------------------------------------
__global__ void rmsnorm_kernel(const float* __restrict__ x,
                               const float* __restrict__ w,
                               float* __restrict__ out, int rows) {
  int wid  = (blockIdx.x * blockDim.x + threadIdx.x) >> 5;
  int lane = threadIdx.x & 31;
  if (wid >= rows) return;
  const float* xr = x + (size_t)wid * H;
  float v[4];
  float ss = 0.0f;
#pragma unroll
  for (int j = 0; j < 4; ++j) { v[j] = xr[lane + 32 * j]; ss += v[j] * v[j]; }
#pragma unroll
  for (int m = 16; m >= 1; m >>= 1) ss += __shfl_xor(ss, m, 32);
  float inv = rsqrtf(ss * (1.0f / (float)H) + 1e-5f);
  float* orow = out + (size_t)wid * H;
#pragma unroll
  for (int j = 0; j < 4; ++j) orow[lane + 32 * j] = v[j] * inv * w[lane + 32 * j];
}

// ---------------------------------------------------------------------------
// Weight pack: W[K,N] f32 row-major  ->  Bt[NP,K] f16 (transposed, zero-pad),
// bias[N] -> biasP[NP] (zero-pad). Makes every GEMM B-fragment two contiguous
// 16B global_load_b128s with no guards and no in-loop converts.
// ---------------------------------------------------------------------------
__global__ void pack_weight(const float* __restrict__ W,
                            const float* __restrict__ bias,
                            _Float16* __restrict__ Bt,
                            float* __restrict__ biasP,
                            int K, int N, int NP) {
  int idx = blockIdx.x * blockDim.x + threadIdx.x;
  int total = NP * K;
  if (idx < total) {
    int n = idx / K;
    int k = idx - n * K;
    Bt[idx] = (n < N) ? (_Float16)W[(size_t)k * N + n] : (_Float16)0.0f;
  }
  if (idx < NP) biasP[idx] = (idx < N) ? bias[idx] : 0.0f;
}

// ---------------------------------------------------------------------------
// WMMA GEMM: D[M,NP] = A[M,K](f32) * Bt[NP,K](f16,pre-transposed)
//            + biasP[NP]  (+ addsrc[M,NP] if HAS_ADD)
// Wave computes a 16x64 strip: one A fragment feeds 4 v_wmma_f32_16x16x32_f16.
// All loads for a K-step are issued before any consumer so the scheduler can
// stagger s_wait_loadcnt across the 4 WMMAs. 8 waves / 256-thread block.
// Fragment layouts per CDNA5 ISA 7.12.2 (wave32).
// ---------------------------------------------------------------------------
template <bool HAS_ADD>
__global__ void gemm_wmma_f16(const float* __restrict__ A,
                              const _Float16* __restrict__ Bt,
                              const float* __restrict__ biasP,
                              const float* __restrict__ addsrc,
                              float* __restrict__ D,
                              int M, int NP, int K) {
  const int wave = threadIdx.x >> 5;
  const int lane = threadIdx.x & 31;
  const int ng   = NP >> 6;                 // # of 64-wide column groups
  const int unit = blockIdx.x * 8 + wave;   // wave work unit
  const int tm   = unit / ng;
  const int tn0  = (unit % ng) * 64;
  if (tm * 16 >= M) return;

  const int r16 = lane & 15;   // A: row within tile; B/D: col within tile
  const int kh  = lane >> 4;   // k-half / m-half selector

  v8f acc[4];
#pragma unroll
  for (int c = 0; c < 4; ++c) {
    float bv = biasP[tn0 + c * 16 + r16];
#pragma unroll
    for (int r = 0; r < 8; ++r) acc[c][r] = bv;
  }

  const float* arow = A + (size_t)(tm * 16 + r16) * K;
  for (int k0 = 0; k0 < K; k0 += 32) {
    // --- issue all loads for this K-step first -----------------------------
    v8h blo[4], bhi[4];
#pragma unroll
    for (int c = 0; c < 4; ++c) {
      const _Float16* bcol =
          Bt + (size_t)(tn0 + c * 16 + r16) * K + k0 + kh * 8;
      blo[c] = *(const v8h*)bcol;           // K = k0+kh*8     .. +8  (16B)
      bhi[c] = *(const v8h*)(bcol + 16);    // K = k0+16+kh*8  .. +8  (16B)
    }
    v4f a0 = *(const v4f*)(arow + k0 + kh * 8);
    v4f a1 = *(const v4f*)(arow + k0 + kh * 8 + 4);
    v4f a2 = *(const v4f*)(arow + k0 + 16 + kh * 8);
    v4f a3 = *(const v4f*)(arow + k0 + 16 + kh * 8 + 4);

    // --- convert A, then 4 back-to-back WMMAs ------------------------------
    v16h af;
#pragma unroll
    for (int j = 0; j < 4; ++j) {
      af[j]      = (_Float16)a0[j];
      af[j + 4]  = (_Float16)a1[j];
      af[j + 8]  = (_Float16)a2[j];
      af[j + 12] = (_Float16)a3[j];
    }
#pragma unroll
    for (int c = 0; c < 4; ++c) {
      v16h bf = __builtin_shufflevector(blo[c], bhi[c],
                  0, 1, 2, 3, 4, 5, 6, 7, 8, 9, 10, 11, 12, 13, 14, 15);
      acc[c] = __builtin_amdgcn_wmma_f32_16x16x32_f16(
          /*neg_a=*/false, af, /*neg_b=*/false, bf,
          /*c_mod=*/(short)0, acc[c], /*reuse_a=*/false, /*reuse_b=*/false);
    }
  }

#pragma unroll
  for (int c = 0; c < 4; ++c) {
    int coln = tn0 + c * 16 + r16;
#pragma unroll
    for (int r = 0; r < 8; ++r) {
      int row    = tm * 16 + kh * 8 + r;    // D: vgpr r -> M = r + 8*kh
      size_t idx = (size_t)row * NP + coln;
      if (HAS_ADD)
        D[idx] = acc[c][r] + addsrc[idx];
      else
        D[idx] = acc[c][r];
    }
  }
}

// ---------------------------------------------------------------------------
// Causal depthwise conv (K=4) + SiLU. Reads xc channels out of xz (stride 2E).
// ---------------------------------------------------------------------------
__global__ void conv_silu_kernel(const float* __restrict__ xz,
                                 const float* __restrict__ cw,
                                 const float* __restrict__ cb,
                                 float* __restrict__ xc) {
  int idx = blockIdx.x * blockDim.x + threadIdx.x;
  if (idx >= BATCH * SEQ * E) return;
  int d = idx & (E - 1);
  int l = (idx >> 8) & (SEQ - 1);
  int b = idx >> 18;
  float acc = cb[d];
#pragma unroll
  for (int j = 0; j < KCONV; ++j) {
    int li = l - (KCONV - 1) + j;
    if (li >= 0)
      acc += cw[d * KCONV + j] * xz[((size_t)b * SEQ + li) * (2 * E) + d];
  }
  xc[idx] = acc / (1.0f + __expf(-acc));   // silu
}

// ---------------------------------------------------------------------------
// delta = softplus(dbc[:, :R] @ dt_w + dt_b)   (dbc has padded stride DBC_P)
// ---------------------------------------------------------------------------
__global__ void delta_kernel(const float* __restrict__ dbc,
                             const float* __restrict__ dt_w,
                             const float* __restrict__ dt_b,
                             float* __restrict__ delta) {
  int idx = blockIdx.x * blockDim.x + threadIdx.x;
  if (idx >= BATCH * SEQ * E) return;
  int d   = idx & (E - 1);
  int row = idx >> 8;
  float v = dt_b[d];
#pragma unroll
  for (int r = 0; r < R; ++r) v += dbc[(size_t)row * DBC_P + r] * dt_w[r * E + d];
  delta[idx] = (v > 20.0f) ? v : __logf(1.0f + __expf(v));
}

// ---------------------------------------------------------------------------
// Selective scan: 1 block per batch, 1024 threads. Thread owns d = tid/4 and
// a 32-wide n-slice; h and -exp(A_log) rows persist in VGPRs for all 1024
// steps. Step t+1 operands are double-buffered into LDS with
// global_load_async_to_lds_b128 while step t computes; s_wait_asynccnt 0 +
// barrier at iteration end. Staging lanes are whole waves (boundaries at
// multiples of 32) so EXEC stays wave-uniform.
// ---------------------------------------------------------------------------
__global__ void scan_kernel(const float* __restrict__ dbc,
                            const float* __restrict__ delta,
                            const float* __restrict__ xc,
                            const float* __restrict__ xz,
                            const float* __restrict__ A_log,
                            const float* __restrict__ Dp,
                            float* __restrict__ yg) {
  const int b   = blockIdx.x;
  const int tid = threadIdx.x;
  const int d   = tid >> 2;
  const int q   = tid & 3;
  const int n0  = q * 32;

  __shared__ __align__(16) float sB[2][N_ST];
  __shared__ __align__(16) float sC[2][N_ST];
  __shared__ __align__(16) float sDl[2][E];
  __shared__ __align__(16) float sX[2][E];

  float h[32], Ar[32];
#pragma unroll
  for (int j = 0; j < 32; ++j) {
    h[j]  = 0.0f;
    Ar[j] = -__expf(A_log[(size_t)d * N_ST + n0 + j]);
  }
  const float dcoef = Dp[d];

  // stage operands for step t into LDS buffer buf (one b128 per lane,
  // waves 0..5 participate; branch boundaries are wave-uniform)
  auto stage = [&](int t, int buf) {
    size_t rbase = (size_t)b * SEQ + t;
    if (tid < 32) {
      async_load_b128_to_lds(dbc + rbase * DBC_P + R + tid * 4,
                             &sB[buf][tid * 4]);
    } else if (tid < 64) {
      int i = tid - 32;
      async_load_b128_to_lds(dbc + rbase * DBC_P + R + N_ST + i * 4,
                             &sC[buf][i * 4]);
    } else if (tid < 128) {
      int i = tid - 64;
      async_load_b128_to_lds(delta + rbase * E + i * 4, &sDl[buf][i * 4]);
    } else if (tid < 192) {
      int i = tid - 128;
      async_load_b128_to_lds(xc + rbase * E + i * 4, &sX[buf][i * 4]);
    }
  };

  stage(0, 0);
  wait_asynccnt0();
  __syncthreads();

  for (int t = 0; t < SEQ; ++t) {
    const int cur = t & 1;
    if (t + 1 < SEQ) stage(t + 1, cur ^ 1);   // prefetch next step async

    float dl  = sDl[cur][d];
    float xv  = sX[cur][d];
    float dx  = dl * xv;
    float acc = 0.0f;
#pragma unroll
    for (int j = 0; j < 32; ++j) {
      float dA = __expf(dl * Ar[j]);
      h[j] = dA * h[j] + dx * sB[cur][n0 + j];
      acc += h[j] * sC[cur][n0 + j];
    }
    acc += __shfl_xor(acc, 1, 32);
    acc += __shfl_xor(acc, 2, 32);
    size_t rbase = (size_t)b * SEQ + t;
    if (q == 0) {
      float s = xz[rbase * (2 * E) + E + d];
      yg[rbase * E + d] = (acc + dcoef * xv) * (s / (1.0f + __expf(-s)));
    }
    wait_asynccnt0();
    __syncthreads();
  }
}

// ---------------------------------------------------------------------------
// Host-side orchestration.
// ---------------------------------------------------------------------------
extern "C" void kernel_launch(void* const* d_in, const int* in_sizes, int n_in,
                              void* d_out, int out_size, void* d_ws, size_t ws_size,
                              hipStream_t stream) {
  (void)in_sizes; (void)n_in; (void)out_size; (void)ws_size;
  const float* x       = (const float*)d_in[0];
  const float* in_w    = (const float*)d_in[1];
  const float* in_b    = (const float*)d_in[2];
  const float* out_w   = (const float*)d_in[3];
  const float* out_b   = (const float*)d_in[4];
  const float* pl_w    = (const float*)d_in[5];
  const float* pl_b    = (const float*)d_in[6];
  const float* dt_w    = (const float*)d_in[7];
  const float* dt_b    = (const float*)d_in[8];
  const float* conv_w  = (const float*)d_in[9];
  const float* conv_b  = (const float*)d_in[10];
  const float* A_log   = (const float*)d_in[11];
  const float* Dp      = (const float*)d_in[12];
  const float* norm_w  = (const float*)d_in[13];
  const float* fnorm_w = (const float*)d_in[14];

  const int ROWS   = BATCH * SEQ;               // 2048
  const int NP_IN  = 2 * E;                     // 512
  const int NP_PL  = DBC_P;                     // 320
  const int NP_OUT = H;                         // 128

  float* ws    = (float*)d_ws;
  float* x_cur = ws;  ws += (size_t)ROWS * H;        // 262144
  float* xn    = ws;  ws += (size_t)ROWS * H;        // 262144
  float* xz    = ws;  ws += (size_t)ROWS * 2 * E;    // 1048576
  float* xc    = ws;  ws += (size_t)ROWS * E;        // 524288
  float* dbc   = ws;  ws += (size_t)ROWS * DBC_P;    // 655360
  float* delta = ws;  ws += (size_t)ROWS * E;        // 524288
  float* yg    = ws;  ws += (size_t)ROWS * E;        // 524288
  float* biasA = ws;  ws += NP_IN;
  float* biasB = ws;  ws += NP_PL;
  float* biasC = ws;  ws += NP_OUT;
  _Float16* wtA = (_Float16*)ws;                     // 512*128 f16
  _Float16* wtB = wtA + (size_t)NP_IN * H;           // 320*256 f16
  _Float16* wtC = wtB + (size_t)NP_PL * E;           // 128*256 f16

  const float* xin = x;
  for (int layer = 0; layer < 2; ++layer) {
    const float* l_in_w  = in_w   + (size_t)layer * H * 2 * E;
    const float* l_in_b  = in_b   + (size_t)layer * 2 * E;
    const float* l_out_w = out_w  + (size_t)layer * E * H;
    const float* l_out_b = out_b  + (size_t)layer * H;
    const float* l_pl_w  = pl_w   + (size_t)layer * E * DBC_N;
    const float* l_pl_b  = pl_b   + (size_t)layer * DBC_N;
    const float* l_dt_w  = dt_w   + (size_t)layer * R * E;
    const float* l_dt_b  = dt_b   + (size_t)layer * E;
    const float* l_cw    = conv_w + (size_t)layer * E * KCONV;
    const float* l_cb    = conv_b + (size_t)layer * E;
    const float* l_A     = A_log  + (size_t)layer * E * N_ST;
    const float* l_D     = Dp     + (size_t)layer * E;
    const float* l_nw    = norm_w + (size_t)layer * H;

    // 0. pack weights -> f16 transposed, padded
    pack_weight<<<(NP_IN * H + 255) / 256, 256, 0, stream>>>(
        l_in_w, l_in_b, wtA, biasA, H, 2 * E, NP_IN);
    pack_weight<<<(NP_PL * E + 255) / 256, 256, 0, stream>>>(
        l_pl_w, l_pl_b, wtB, biasB, E, DBC_N, NP_PL);
    pack_weight<<<(NP_OUT * E + 255) / 256, 256, 0, stream>>>(
        l_out_w, l_out_b, wtC, biasC, E, H, NP_OUT);

    // 1. rmsnorm -> xn
    rmsnorm_kernel<<<ROWS * 32 / 256, 256, 0, stream>>>(xin, l_nw, xn, ROWS);

    // 2. in_proj: xz = xn @ in_w + in_b   (M=2048, NP=512, K=128)
    {
      int units = (ROWS / 16) * (NP_IN / 64);     // 1024
      gemm_wmma_f16<false><<<units / 8, 256, 0, stream>>>(
          xn, wtA, biasA, nullptr, xz, ROWS, NP_IN, H);
    }

    // 3. causal conv + silu -> xc
    conv_silu_kernel<<<(ROWS * E) / 256, 256, 0, stream>>>(xz, l_cw, l_cb, xc);

    // 4. pl_proj: dbc = xc @ pl_w + pl_b  (M=2048, NP=320, K=256)
    {
      int units = (ROWS / 16) * (NP_PL / 64);     // 640
      gemm_wmma_f16<false><<<units / 8, 256, 0, stream>>>(
          xc, wtB, biasB, nullptr, dbc, ROWS, NP_PL, E);
    }

    // 5. delta = softplus(dbc[:, :R] @ dt_w + dt_b)
    delta_kernel<<<(ROWS * E) / 256, 256, 0, stream>>>(dbc, l_dt_w, l_dt_b, delta);

    // 6. selective scan + D*x + silu(skip) gating -> yg
    scan_kernel<<<BATCH, 1024, 0, stream>>>(dbc, delta, xc, xz, l_A, l_D, yg);

    // 7. out_proj + residual: x_cur = yg @ out_w + out_b + xin
    {
      int units = (ROWS / 16) * (NP_OUT / 64);    // 256
      gemm_wmma_f16<true><<<units / 8, 256, 0, stream>>>(
          yg, wtC, biasC, xin, x_cur, ROWS, NP_OUT, E);
    }
    xin = x_cur;
  }

  // final rmsnorm -> d_out
  rmsnorm_kernel<<<ROWS * 32 / 256, 256, 0, stream>>>(x_cur, fnorm_w,
                                                      (float*)d_out, ROWS);
}
